// MultiHeadAttention_89129161327077
// MI455X (gfx1250) — matrix operand order repaired and checked by reading, not compiled, not run on hardware
//
#include <hip/hip_runtime.h>
#include <hip/hip_bf16.h>

// MI455X / gfx1250 flash multi-head attention, f16 WMMA with f32 accumulation.
// B=2, S=2048, D=1024, H=16, HD=64.

typedef __attribute__((ext_vector_type(16))) _Float16 v16h;
typedef __attribute__((ext_vector_type(8)))  _Float16 v8h;
typedef __attribute__((ext_vector_type(8)))  float    v8f;
typedef int v4i_gcc __attribute__((vector_size(16)));   // matches builtin's V4i

#define Bb 2
#define Ss 2048
#define Dd 1024
#define Hh 16
#define HD 64
#define KT 32                 // keys per flash-attention k-tile
#define Mrows (Bb * Ss)       // 4096
#define LOG2E 1.44269504088896340736f

// ---- optional gfx1250 async global->LDS path (probe via __has_builtin) ----
#if defined(__has_builtin)
#if __has_builtin(__builtin_amdgcn_global_load_async_to_lds_b128)
#define HAVE_ASYNC_LDS 1
#endif
#endif

// builtin signature (from hipcc diagnostic): param0 = AS1 v4i*, param1 = AS3 v4i*
#define AS1CAST(p) ((__attribute__((address_space(1))) v4i_gcc*)(p))
#define AS3CAST(p) ((__attribute__((address_space(3))) v4i_gcc*)(p))

__device__ __forceinline__ void wait_async_lds() {
#if defined(HAVE_ASYNC_LDS)
#if __has_builtin(__builtin_amdgcn_s_wait_asynccnt)
  __builtin_amdgcn_s_wait_asynccnt(0);
#else
  asm volatile("s_wait_asynccnt 0x0" ::: "memory");
#endif
#endif
}

// ---------------- fragment loaders (contiguous k = hs*16 + j for A and B) ----------------
// A 16x32 f16: lane L (l16=L&15, hs=L>>4): row M=l16; halves j = K[kOff + j],
// kOff = hs*16.  p must already include kOff.
__device__ __forceinline__ v16h load_a16(const _Float16* p) {
  v8h lo = *(const v8h*)p;
  v8h hi = *(const v8h*)(p + 8);
  v16h r;
#pragma unroll
  for (int i = 0; i < 8; ++i) { r[i] = lo[i]; r[8 + i] = hi[i]; }
  return r;
}
// B 32x16 f16: lane L: col N=l16; halves j = K[kb + j], kb = hs*16. p includes kb.
__device__ __forceinline__ v16h load_b16(const _Float16* p) {
  v8h lo = *(const v8h*)p;
  v8h hi = *(const v8h*)(p + 8);
  v16h r;
#pragma unroll
  for (int i = 0; i < 8; ++i) { r[i] = lo[i]; r[8 + i] = hi[i]; }
  return r;
}

__device__ __forceinline__ v8f wmma_f16(v16h a, v16h b, v8f c) {
  return __builtin_amdgcn_wmma_f32_16x16x32_f16(false, a, false, b, (short)0, c,
                                                false, false);
}

// 16-lane-group reductions via ds_swizzle (xor masks stay inside 16-lane halves,
// matching the WMMA C layout: lanes 0-15 = rows 0-7, lanes 16-31 = rows 8-15).
#define SWZ(v, m) \
  __int_as_float(__builtin_amdgcn_ds_swizzle(__float_as_int(v), (0x1F | ((m) << 10))))
__device__ __forceinline__ float rmax16(float v) {
  v = fmaxf(v, SWZ(v, 1)); v = fmaxf(v, SWZ(v, 2));
  v = fmaxf(v, SWZ(v, 4)); v = fmaxf(v, SWZ(v, 8));
  return v;
}
__device__ __forceinline__ float rsum16(float v) {
  v += SWZ(v, 1); v += SWZ(v, 2); v += SWZ(v, 4); v += SWZ(v, 8);
  return v;
}

// ---------------- shared GEMM core: 32x64 wave tile, K = Dd ----------------
// out[m][n] = sum_k A[m][k] * B[n][k]   (A rows at ap0/ap1, B rows at bp)
// Ping-pong double buffering: even/odd fragment sets are loaded directly into
// their own registers -> no rotation copies (v_mov) in the steady-state loop.
__device__ __forceinline__ void gemm_32x64(const _Float16* __restrict__ ap0,
                                           const _Float16* __restrict__ ap1,
                                           const _Float16* __restrict__ bp,
                                           v8f c0[4], v8f c1[4]) {
  v16h a0e, a1e, be[4];           // even k-chunk fragments
  a0e = load_a16(ap0);
  a1e = load_a16(ap1);
#pragma unroll
  for (int t = 0; t < 4; ++t) be[t] = load_b16(bp + t * 16 * Dd);

  for (int kk = 0; kk + 64 < Dd; kk += 64) {
    // odd chunk loads (kk+32), then consume even fragments
    v16h a0o = load_a16(ap0 + kk + 32);
    v16h a1o = load_a16(ap1 + kk + 32);
    v16h bo[4];
#pragma unroll
    for (int t = 0; t < 4; ++t) bo[t] = load_b16(bp + t * 16 * Dd + kk + 32);
#pragma unroll
    for (int t = 0; t < 4; ++t) {
      c0[t] = wmma_f16(a0e, be[t], c0[t]);
      c1[t] = wmma_f16(a1e, be[t], c1[t]);
    }
    // next even chunk loads (kk+64), then consume odd fragments
    a0e = load_a16(ap0 + kk + 64);
    a1e = load_a16(ap1 + kk + 64);
#pragma unroll
    for (int t = 0; t < 4; ++t) be[t] = load_b16(bp + t * 16 * Dd + kk + 64);
#pragma unroll
    for (int t = 0; t < 4; ++t) {
      c0[t] = wmma_f16(a0o, bo[t], c0[t]);
      c1[t] = wmma_f16(a1o, bo[t], c1[t]);
    }
  }
  // tail: chunks Dd-64 (already in even regs) and Dd-32
  {
    v16h a0o = load_a16(ap0 + Dd - 32);
    v16h a1o = load_a16(ap1 + Dd - 32);
    v16h bo[4];
#pragma unroll
    for (int t = 0; t < 4; ++t) bo[t] = load_b16(bp + t * 16 * Dd + Dd - 32);
#pragma unroll
    for (int t = 0; t < 4; ++t) {
      c0[t] = wmma_f16(a0e, be[t], c0[t]);
      c1[t] = wmma_f16(a1e, be[t], c1[t]);
    }
#pragma unroll
    for (int t = 0; t < 4; ++t) {
      c0[t] = wmma_f16(a0o, bo[t], c0[t]);
      c1[t] = wmma_f16(a1o, bo[t], c1[t]);
    }
  }
}

// ---------------- kernel 1: f32 -> f16 convert ----------------
__global__ void __launch_bounds__(256) cvt_f32_to_f16(const float* __restrict__ src,
                                                      _Float16* __restrict__ dst, int n) {
  int i = blockIdx.x * blockDim.x + threadIdx.x;
  int stride = gridDim.x * blockDim.x;
  for (; i < n; i += stride) dst[i] = (_Float16)src[i];
}

// ---------------- kernel 2: QKV projection ----------------
// grid (M/256, N/64, 3), block 256 (8 waves).  Q,K -> [B,H,S,HD]; V -> [B,H,HD,S].
__global__ void __launch_bounds__(256) qkv_proj(
    const _Float16* __restrict__ x,
    const _Float16* __restrict__ Wq16, const _Float16* __restrict__ Wk16,
    const _Float16* __restrict__ Wv16,
    const float* __restrict__ bq, const float* __restrict__ bk,
    const float* __restrict__ bv,
    _Float16* __restrict__ Qf, _Float16* __restrict__ Kf,
    _Float16* __restrict__ Vt) {
  const int lane = threadIdx.x & 31;
  const int wave = threadIdx.x >> 5;
  const int l16 = lane & 15;
  const int hs = lane >> 4;
  const int mTile = blockIdx.x * 256 + wave * 32;
  const int nTile = blockIdx.y * 64;     // head = blockIdx.y
  const int which = blockIdx.z;

  const _Float16* W16 = (which == 0) ? Wq16 : (which == 1) ? Wk16 : Wv16;
  const float* bias = (which == 0) ? bq : (which == 1) ? bk : bv;

  const _Float16* ap0 = x + (mTile + l16) * Dd + hs * 16;
  const _Float16* ap1 = ap0 + 16 * Dd;
  const _Float16* bp  = W16 + (nTile + l16) * Dd + hs * 16;

  v8f c0[4] = {}, c1[4] = {};
  gemm_32x64(ap0, ap1, bp, c0, c1);

  const int rowOff = hs * 8;
  const int head = blockIdx.y;
#pragma unroll
  for (int t = 0; t < 4; ++t) {
    const int n = nTile + t * 16 + l16;
    const int hd = t * 16 + l16;
    const float bval = bias[n];
#pragma unroll
    for (int half = 0; half < 2; ++half) {
      const v8f* cc = half ? c1 : c0;
#pragma unroll
      for (int r = 0; r < 8; ++r) {
        const int m = mTile + half * 16 + rowOff + r;
        const int bIdx = m >> 11;          // / S
        const int sIdx = m & (Ss - 1);
        const int bh = bIdx * Hh + head;
        const _Float16 hv = (_Float16)(cc[t][r] + bval);
        if (which == 2)
          Vt[(bh * HD + hd) * Ss + sIdx] = hv;        // transposed V
        else if (which == 0)
          Qf[(bh * Ss + sIdx) * HD + hd] = hv;
        else
          Kf[(bh * Ss + sIdx) * HD + hd] = hv;
      }
    }
  }
}

// ---------------- kernel 3: causal flash attention ----------------
// Block = 8 waves, all on the SAME head, q-tiles qt = group*8 + wave.
// K (32x64) and V^T (64x32) tiles are staged cooperatively into LDS with
// double-buffered async global->LDS copies shared by all 8 waves.
// grid = (B*H) * (S/16/8) = 512 blocks.
__global__ void __launch_bounds__(256) flash_attn(
    const _Float16* __restrict__ Qf, const _Float16* __restrict__ Kf,
    const _Float16* __restrict__ Vt, _Float16* __restrict__ Of) {
  __shared__ __align__(16) _Float16 ldsK[2][KT * HD];   // 2 x 4 KB
  __shared__ __align__(16) _Float16 ldsV[2][HD * KT];   // 2 x 4 KB
  __shared__ __align__(16) _Float16 ldsP[8][16 * 32];   // 8 KB (per-wave P tile)

  const int tid = threadIdx.x;
  const int lane = tid & 31;
  const int wave = tid >> 5;
  const int l16 = lane & 15;
  const int hs = lane >> 4;
  const int rowOff = hs * 8;

  const int bh = blockIdx.x >> 4;                 // 0..31
  const int group = blockIdx.x & 15;
  const int qt = group * 8 + wave;                // 0..127

  const _Float16* Qbase = Qf + (bh * Ss + qt * 16) * HD;
  const _Float16* Kbase = Kf + bh * Ss * HD;
  const _Float16* Vbase = Vt + bh * HD * Ss;

  // Q fragments stay in registers for the whole k-loop.
  const v16h qa0 = load_a16(Qbase + l16 * HD + 0  + hs * 16);
  const v16h qa1 = load_a16(Qbase + l16 * HD + 32 + hs * 16);

  v8f acc[4] = {};
  float mrow[8], lrow[8];
#pragma unroll
  for (int r = 0; r < 8; ++r) { mrow[r] = -1e30f; lrow[r] = 0.f; }

  _Float16* P = ldsP[wave];
  const int ktEnd = (qt * 16 + 15) >> 5;                // per-wave causal bound
  const int ktEndMax = ((group * 8 + 7) * 16 + 15) >> 5; // block-wide bound

  // cooperative tile staging: 256 threads x one b128 each per tile
  const int krow = tid >> 3, kchk = tid & 7;   // K tile rows 0..31, 8 chunks of 8 halfs
  const int vrow = tid >> 2, vchk = tid & 3;   // V tile rows 0..63, 4 chunks of 8 halfs
  auto stage = [&](int kt, int buf) {
    const _Float16* gk = Kbase + (kt * KT + krow) * HD + kchk * 8;
    const _Float16* gv = Vbase + vrow * Ss + kt * KT + vchk * 8;
    _Float16* lk = &ldsK[buf][krow * HD + kchk * 8];
    _Float16* lv = &ldsV[buf][vrow * KT + vchk * 8];
#if defined(HAVE_ASYNC_LDS)
    __builtin_amdgcn_global_load_async_to_lds_b128(AS1CAST(gk), AS3CAST(lk), 0, 0);
    __builtin_amdgcn_global_load_async_to_lds_b128(AS1CAST(gv), AS3CAST(lv), 0, 0);
#else
    *(v8h*)lk = *(const v8h*)gk;
    *(v8h*)lv = *(const v8h*)gv;
#endif
  };

  stage(0, 0);
  for (int kt = 0; kt <= ktEndMax; ++kt) {
    const int buf = kt & 1;
    wait_async_lds();        // my async copies into ldsX[buf] have landed
    __syncthreads();         // everyone's copies visible; prior reads retired
    if (kt < ktEndMax) stage(kt + 1, buf ^ 1);

    if (kt <= ktEnd) {       // wave-uniform: EXEC stays all-ones for WMMA
      const _Float16* Kt_ = ldsK[buf];
      const _Float16* Vt_ = ldsV[buf];
      const int kBase = kt * KT;

      const v16h kb00 = load_b16(Kt_ + (0  + l16) * HD + 0  + hs * 16);
      const v16h kb01 = load_b16(Kt_ + (0  + l16) * HD + 32 + hs * 16);
      const v16h kb10 = load_b16(Kt_ + (16 + l16) * HD + 0  + hs * 16);
      const v16h kb11 = load_b16(Kt_ + (16 + l16) * HD + 32 + hs * 16);

      v8f s0 = {}, s1 = {};
      s0 = wmma_f16(qa0, kb00, s0);
      s0 = wmma_f16(qa1, kb01, s0);
      s1 = wmma_f16(qa0, kb10, s1);
      s1 = wmma_f16(qa1, kb11, s1);

      const bool needMask = (kBase + 31) > (qt * 16);
      const int ncol = kBase + l16;

#pragma unroll
      for (int r = 0; r < 8; ++r) {
        float v0 = s0[r] * 0.125f;               // 1/sqrt(64)
        float v1 = s1[r] * 0.125f;
        if (needMask) {
          const int qrow = qt * 16 + rowOff + r;
          if (ncol      > qrow) v0 = -1e30f;
          if (ncol + 16 > qrow) v1 = -1e30f;
        }
        const float tmax = rmax16(fmaxf(v0, v1));
        const float mnew = fmaxf(mrow[r], tmax);
        const float alpha = __builtin_amdgcn_exp2f((mrow[r] - mnew) * LOG2E);
        const float p0 = __builtin_amdgcn_exp2f((v0 - mnew) * LOG2E);
        const float p1 = __builtin_amdgcn_exp2f((v1 - mnew) * LOG2E);
        lrow[r] = lrow[r] * alpha + rsum16(p0 + p1);
        mrow[r] = mnew;
        acc[0][r] *= alpha; acc[1][r] *= alpha;
        acc[2][r] *= alpha; acc[3][r] *= alpha;
        // C-layout -> LDS (row-major P[16][32]) for A-layout reload
        P[(rowOff + r) * 32 + l16]      = (_Float16)p0;
        P[(rowOff + r) * 32 + 16 + l16] = (_Float16)p1;
      }

      const v16h pa = load_a16(P + l16 * 32 + hs * 16);  // P as A-frag (16x32)
#pragma unroll
      for (int t = 0; t < 4; ++t) {
        const v16h vb = load_b16(Vt_ + (t * 16 + l16) * KT + hs * 16);
        acc[t] = wmma_f16(pa, vb, acc[t]);
      }
    }
  }

  // normalize, store O as f16 in [B,S,H*HD] (matches transpose+reshape)
  const int b = bh >> 4;
  const int head = bh & 15;
#pragma unroll
  for (int r = 0; r < 8; ++r) {
    const float inv = __builtin_amdgcn_rcpf(lrow[r]);
    const int qrow = qt * 16 + rowOff + r;
    const int mrowg = b * Ss + qrow;
#pragma unroll
    for (int t = 0; t < 4; ++t)
      Of[mrowg * Dd + head * HD + t * 16 + l16] = (_Float16)(acc[t][r] * inv);
  }
}

// ---------------- kernel 4: output projection (f32 out) ----------------
// Same pipelined 32x64 wave tile; grid (M/256, N/64).
__global__ void __launch_bounds__(256) out_proj(
    const _Float16* __restrict__ x, const _Float16* __restrict__ Wo16,
    const float* __restrict__ bo, float* __restrict__ out) {
  const int lane = threadIdx.x & 31;
  const int wave = threadIdx.x >> 5;
  const int l16 = lane & 15;
  const int hs = lane >> 4;
  const int mTile = blockIdx.x * 256 + wave * 32;
  const int nTile = blockIdx.y * 64;

  const _Float16* ap0 = x + (mTile + l16) * Dd + hs * 16;
  const _Float16* ap1 = ap0 + 16 * Dd;
  const _Float16* bp  = Wo16 + (nTile + l16) * Dd + hs * 16;

  v8f c0[4] = {}, c1[4] = {};
  gemm_32x64(ap0, ap1, bp, c0, c1);

  const int rowOff = hs * 8;
#pragma unroll
  for (int t = 0; t < 4; ++t) {
    const int n = nTile + t * 16 + l16;
    const float bval = bo[n];
#pragma unroll
    for (int r = 0; r < 8; ++r) {
      out[(mTile + rowOff + r) * Dd + n] = c0[t][r] + bval;
      out[(mTile + 16 + rowOff + r) * Dd + n] = c1[t][r] + bval;
    }
  }
}

// ---------------- host launcher ----------------
extern "C" void kernel_launch(void* const* d_in, const int* in_sizes, int n_in,
                              void* d_out, int out_size, void* d_ws, size_t ws_size,
                              hipStream_t stream) {
  (void)in_sizes; (void)n_in; (void)out_size; (void)ws_size;
  const float* h  = (const float*)d_in[0];
  const float* Wq = (const float*)d_in[1];
  const float* bq = (const float*)d_in[2];
  const float* Wk = (const float*)d_in[3];
  const float* bk = (const float*)d_in[4];
  const float* Wv = (const float*)d_in[5];
  const float* bv = (const float*)d_in[6];
  const float* Wo = (const float*)d_in[7];
  const float* bo = (const float*)d_in[8];

  char* ws = (char*)d_ws;
  const size_t MB = 1024ull * 1024ull;
  _Float16* hf   = (_Float16*)(ws + 0);          // 8 MB  [4096][1024]
  _Float16* Wq16 = (_Float16*)(ws + 8 * MB);     // 2 MB
  _Float16* Wk16 = (_Float16*)(ws + 10 * MB);    // 2 MB
  _Float16* Wv16 = (_Float16*)(ws + 12 * MB);    // 2 MB
  _Float16* Wo16 = (_Float16*)(ws + 14 * MB);    // 2 MB
  _Float16* Qf   = (_Float16*)(ws + 16 * MB);    // 8 MB  [B,H,S,HD]
  _Float16* Kf   = (_Float16*)(ws + 24 * MB);    // 8 MB  [B,H,S,HD]
  _Float16* Vt   = (_Float16*)(ws + 32 * MB);    // 8 MB  [B,H,HD,S]
  _Float16* Of   = (_Float16*)(ws + 40 * MB);    // 8 MB  [B,S,D]

  cvt_f32_to_f16<<<1024, 256, 0, stream>>>(h,  hf,   Mrows * Dd);
  cvt_f32_to_f16<<<512,  256, 0, stream>>>(Wq, Wq16, Dd * Dd);
  cvt_f32_to_f16<<<512,  256, 0, stream>>>(Wk, Wk16, Dd * Dd);
  cvt_f32_to_f16<<<512,  256, 0, stream>>>(Wv, Wv16, Dd * Dd);
  cvt_f32_to_f16<<<512,  256, 0, stream>>>(Wo, Wo16, Dd * Dd);

  qkv_proj<<<dim3(Mrows / 256, Dd / 64, 3), 256, 0, stream>>>(
      hf, Wq16, Wk16, Wv16, bq, bk, bv, Qf, Kf, Vt);

  flash_attn<<<(Bb * Hh) * (Ss / 16 / 8), 256, 0, stream>>>(Qf, Kf, Vt, Of);

  out_proj<<<dim3(Mrows / 256, Dd / 64), 256, 0, stream>>>(
      Of, Wo16, bo, (float*)d_out);
}
